// SparseConvEncoder_33792802685224
// MI455X (gfx1250) — matrix operand order repaired and checked by reading, not compiled, not run on hardware
//
#include <hip/hip_runtime.h>

// ---------------------------------------------------------------------------
// Sparse 3D conv encoder on MI455X (gfx1250, wave32, WMMA).
//   L1: feats[N,1] -> x1[N,16]   (gather-GEMM: 27 offsets = K dim, 1 WMMA/tile)
//   L2: x1[N,16]   -> x2[N,32]   (27-offset gather-GEMM, K=32 = 2 offsets x 16 Cin)
//   FC: x2[N,32]   -> out[N,10]  (fused, 1 WMMA per 16-voxel tile)
// All matrix math via v_wmma_f32_16x16x32_f16. Branchless gathers (sentinel
// row / clamp+mask, 32-bit GVS offsets). B operands staged in LDS.
// l2fc processes 32 voxels per wave: each B fragment feeds 2 WMMAs with the
// reuse chain (a0,b0)->(a0,b1)[RA]->(a1,b1)[RB]->(a1,b0)[RA].
// Stores use wave-uniform tile guards (scalar branch) + a single lane-mask
// region, avoiding per-store exec churn next to the WMMAs.
// ---------------------------------------------------------------------------

typedef __attribute__((ext_vector_type(16))) _Float16 v16h;
typedef __attribute__((ext_vector_type(8)))  _Float16 v8h;
typedef __attribute__((ext_vector_type(8)))  float    v8f;

union AFrag  { v16h v; v8h h[2]; };
union AElems { v16h v; _Float16 e[16]; };

#define NOFF 14                   // 27 kernel offsets padded to 14 pairs (K=32)
#define BW2_HALFS (14*2*32*16)    // [pair][couthalf][lane][16 halfs] = 14336
#define BFC_HALFS (32*16)         // [lane][16 halfs] = 512
#define BW1_HALFS (32*16)         // [lane][16 halfs] = 512

#define BW2_OFF  0
#define BFC_OFF  (BW2_HALFS)                  // 14336
#define BW1_OFF  (BW2_HALFS + BFC_HALFS)      // 14848
#define SENT_OFF (BW1_OFF + BW1_HALFS)        // 15360

// ---------------------------------------------------------------------------
// Prep: shuffle W2 [27][16][32], Wfc [32][10], W1 [27][16] (all f32) into the
// per-lane CDNA5 16-bit B-matrix layout; zero-pad offset 27..31 / cols 10..15.
// Also zeroes the gather sentinel row x1[N].
// ---------------------------------------------------------------------------
__global__ __launch_bounds__(256) void prep_kernel(
    const float* __restrict__ W2, const float* __restrict__ Wfc,
    const float* __restrict__ W1,
    _Float16* __restrict__ Bw2, _Float16* __restrict__ Bfc,
    _Float16* __restrict__ Bw1,
    _Float16* __restrict__ x1, int N)
{
    int t = blockIdx.x * blockDim.x + threadIdx.x;
    if (t < BW2_HALFS) {
        int q = t & 15;          // element slot; cin = 8*(L>>4) + (q&7)
        int L = (t >> 4) & 31;   // lane
        int j = (t >> 9) & 1;    // cout half
        int p = t >> 10;         // offset pair
        int k = 2 * p + (q >> 3);
        int c = (L & 15) + 16 * j;
        float v = (k < 27) ? W2[((long)k * 16 + 8 * (L >> 4) + (q & 7)) * 32 + c] : 0.0f;
        Bw2[t] = (_Float16)v;
    } else if (t < BW1_OFF) {
        int s = t - BFC_OFF;
        int q = s & 15;
        int L = s >> 4;
        int K = (q < 8) ? 8 * (L >> 4) + q : 16 + 8 * (L >> 4) + (q - 8);   // K == cout of x2
        int c = L & 15;                  // FC output column
        float v = (c < 10) ? Wfc[K * 10 + c] : 0.0f;
        Bfc[s] = (_Float16)v;
    } else if (t < SENT_OFF) {
        int s = t - BW1_OFF;
        int q = s & 15;
        int L = s >> 4;
        int K = (L < 16) ? q : 16 + q;   // K == kernel offset (pad 27..31)
        int c = L & 15;                  // cout of x1
        float v = (K < 27) ? W1[K * 16 + c] : 0.0f;
        Bw1[s] = (_Float16)v;
    } else if (t < SENT_OFF + 16) {
        x1[(long)N * 16 + (t - SENT_OFF)] = (_Float16)0.0f;
    }
}

// ---------------------------------------------------------------------------
// Layer 1 as gather-GEMM: one wave per 16-voxel tile, one WMMA.
// A[m][k] = feats[kmap[k][m]] (27 offsets padded to K=32). Lane L supplies
// rows M=L&15 with K = h*16..h*16+15 (h=L>>4).
// Branchless: clamp k to 26 / idx to 0 for the loads, mask values to zero.
// ---------------------------------------------------------------------------
__global__ __launch_bounds__(256) void l1_kernel(
    const float* __restrict__ feats, const int* __restrict__ kmap,
    const _Float16* __restrict__ Bw1, _Float16* __restrict__ x1, int N)
{
    const int lane = threadIdx.x & 31;
    const int wid  = threadIdx.x >> 5;
    const int gwave = blockIdx.x * 8 + wid;
    const int base = gwave * 16;
    const int M = lane & 15;
    const int h = lane >> 4;
    int rowM = base + M;
    if (rowM >= N) rowM = 0;                 // clamped; stores guarded below

    AElems a;
#pragma unroll
    for (int s = 0; s < 16; ++s) {
        const int k = h * 16 + s;                              // 16*h+s
        const int kk = k < 27 ? k : 26;                        // clamp (no OOB)
        int idx = kmap[(unsigned)(kk * N) + (unsigned)rowM];
        int j = idx >= 0 ? idx : 0;
        float v = feats[(unsigned)j];
        v = (idx >= 0 && k < 27) ? v : 0.0f;                   // mask
        a.e[s] = (_Float16)v;
    }

    const v16h b = *(const v16h*)(Bw1 + lane * 16);
    v8f c = {};
    c = __builtin_amdgcn_wmma_f32_16x16x32_f16(false, a.v, false, b,
                                               (short)0, c, false, false);

    if (base + 16 <= N) {                    // wave-uniform fast path
#pragma unroll
        for (int r = 0; r < 8; ++r) {
            int row = base + r + 8 * h;
            x1[(unsigned)row * 16u + (unsigned)M] = (_Float16)c[r];
        }
    } else {                                 // tail tile (per-lane guards)
#pragma unroll
        for (int r = 0; r < 8; ++r) {
            int row = base + r + 8 * h;
            if (row < N) x1[(unsigned)row * 16u + (unsigned)M] = (_Float16)c[r];
        }
    }
}

// ---------------------------------------------------------------------------
// Layer 2 + FC: one wave per 32 voxels (two 16-row M-tiles sharing B).
// A fragments assembled in registers from sentinel-guarded b128 gathers
// (32-bit GVS offsets). 56 WMMAs accumulate 2x(16x32), LDS transpose, then
// 2 FC WMMAs (32->10, bias preloaded into C).
// ---------------------------------------------------------------------------
__global__ __launch_bounds__(256) void l2fc_kernel(
    const _Float16* __restrict__ x1,   // [N+1][16] f16 (row N = zeros)
    const int*      __restrict__ kmap, // [27][N]
    const _Float16* __restrict__ Bw2,  // prepped B layout
    const _Float16* __restrict__ Bfc,  // prepped B layout (padded Wfc)
    const float*    __restrict__ bfc,  // [10]
    float*          __restrict__ out,  // [N][10]
    int N)
{
    __shared__ _Float16 sB[BW2_HALFS];      // 28672 B: all W2 B-fragments
    __shared__ _Float16 sBfc[BFC_HALFS];    // 1024 B
    __shared__ _Float16 tile[8][2][16*32];  // per-wave 2x(16x32) staging, 16 KB

    {   // cooperative LDS fill (16B per thread per iter)
        const v8h* gsrc = (const v8h*)Bw2;
        v8h* ldst = (v8h*)sB;
        for (int j = threadIdx.x; j < BW2_HALFS / 8; j += blockDim.x)
            ldst[j] = gsrc[j];
        const v8h* gfc = (const v8h*)Bfc;
        v8h* lfc = (v8h*)sBfc;
        for (int j = threadIdx.x; j < BFC_HALFS / 8; j += blockDim.x)
            lfc[j] = gfc[j];
    }
    __syncthreads();

    const int lane = threadIdx.x & 31;
    const int wid  = threadIdx.x >> 5;
    const int gwave = blockIdx.x * 8 + wid;
    const int base = gwave * 32;
    const int M = lane & 15;
    const int h = lane >> 4;
    int rA = base + M;       if (rA >= N) rA = 0;   // clamped; stores guarded
    int rB = base + 16 + M;  if (rB >= N) rB = 0;
    const unsigned hb = (unsigned)h * 16u;          // byte offset within row
    const char* x1b = (const char*)x1;
    const unsigned uN = (unsigned)N;

    v8f c00 = {}, c01 = {}, c10 = {}, c11 = {};
    const v8h z8 = {};

#pragma unroll
    for (int p = 0; p < NOFF; ++p) {
        const int k0 = 2 * p, k1 = 2 * p + 1;
        const unsigned kb0 = (unsigned)(k0 * N);
        const unsigned kb1 = (unsigned)(k1 * N);

        AFrag a0, a1;
        {
            int i00 = kmap[kb0 + (unsigned)rA];
            int i10 = kmap[kb0 + (unsigned)rB];
            unsigned o0 = (unsigned)(i00 >= 0 ? i00 : (int)uN) * 32u + hb;
            unsigned o1 = (unsigned)(i10 >= 0 ? i10 : (int)uN) * 32u + hb;
            a0.h[0] = *(const v8h*)(x1b + o0);
            a1.h[0] = *(const v8h*)(x1b + o1);
        }
        if (k1 < 27) {                         // compile-time (unrolled)
            int i01 = kmap[kb1 + (unsigned)rA];
            int i11 = kmap[kb1 + (unsigned)rB];
            unsigned o0 = (unsigned)(i01 >= 0 ? i01 : (int)uN) * 32u + hb;
            unsigned o1 = (unsigned)(i11 >= 0 ? i11 : (int)uN) * 32u + hb;
            a0.h[1] = *(const v8h*)(x1b + o0);
            a1.h[1] = *(const v8h*)(x1b + o1);
        } else {
            a0.h[1] = z8;
            a1.h[1] = z8;
        }

        const v16h b0 = *(const v16h*)(sB + p * 1024 + lane * 16);
        const v16h b1 = *(const v16h*)(sB + p * 1024 + 512 + lane * 16);

        c00 = __builtin_amdgcn_wmma_f32_16x16x32_f16(false, a0.v, false, b0,
                                                     (short)0, c00, false, false);
        c01 = __builtin_amdgcn_wmma_f32_16x16x32_f16(false, a0.v, false, b1,
                                                     (short)0, c01, true,  false);
        c11 = __builtin_amdgcn_wmma_f32_16x16x32_f16(false, a1.v, false, b1,
                                                     (short)0, c11, false, true);
        c10 = __builtin_amdgcn_wmma_f32_16x16x32_f16(false, a1.v, false, b0,
                                                     (short)0, c10, true,  false);
    }

    // --- stage both x2 tiles (16 voxels x 32 cout each) into LDS as f16 ---
    _Float16* t0 = tile[wid][0];
    _Float16* t1 = tile[wid][1];
#pragma unroll
    for (int r = 0; r < 8; ++r) {
        int m = r + 8 * h;                 // C layout: vgpr r -> row r (+8 hi)
        t0[m * 32 + M]      = (_Float16)c00[r];
        t0[m * 32 + 16 + M] = (_Float16)c01[r];
        t1[m * 32 + M]      = (_Float16)c10[r];
        t1[m * 32 + 16 + M] = (_Float16)c11[r];
    }
    __syncthreads();

    // --- rebuild FC A-fragments (row M, K chunks per lane half) ---
    AFrag af0, af1;
    af0.h[0] = *(const v8h*)(t0 + M * 32 + h * 8);
    af0.h[1] = *(const v8h*)(t0 + M * 32 + 16 + h * 8);
    af1.h[0] = *(const v8h*)(t1 + M * 32 + h * 8);
    af1.h[1] = *(const v8h*)(t1 + M * 32 + 16 + h * 8);

    const v16h bfcv = *(const v16h*)(sBfc + lane * 16);

    float bias = bfc[M < 10 ? M : 0];      // branchless clamped load + mask
    bias = (M < 10) ? bias : 0.0f;
    v8f cf0, cf1;
#pragma unroll
    for (int r = 0; r < 8; ++r) { cf0[r] = bias; cf1[r] = bias; }

    cf0 = __builtin_amdgcn_wmma_f32_16x16x32_f16(false, af0.v, false, bfcv,
                                                 (short)0, cf0, false, false);
    cf1 = __builtin_amdgcn_wmma_f32_16x16x32_f16(false, af1.v, false, bfcv,
                                                 (short)0, cf1, false, true);

    if (base + 32 <= N) {                  // wave-uniform fast path
        if (M < 10) {                      // single lane-mask region, 16 stores
#pragma unroll
            for (int r = 0; r < 8; ++r) {
                int m = r + 8 * h;
                out[(unsigned)(base + m) * 10u + (unsigned)M]      = cf0[r];
                out[(unsigned)(base + 16 + m) * 10u + (unsigned)M] = cf1[r];
            }
        }
    } else {                               // tail tile (per-lane guards)
#pragma unroll
        for (int r = 0; r < 8; ++r) {
            int m = r + 8 * h;
            int row0 = base + m;
            int row1 = base + 16 + m;
            if (row0 < N && M < 10)
                out[(unsigned)row0 * 10u + (unsigned)M] = cf0[r];
            if (row1 < N && M < 10)
                out[(unsigned)row1 * 10u + (unsigned)M] = cf1[r];
        }
    }
}

// ---------------------------------------------------------------------------
extern "C" void kernel_launch(void* const* d_in, const int* in_sizes, int n_in,
                              void* d_out, int out_size, void* d_ws, size_t ws_size,
                              hipStream_t stream) {
    const float* feats = (const float*)d_in[0];   // [N,1]
    const int*   kmap  = (const int*)d_in[1];     // [27,N]
    const float* W1    = (const float*)d_in[2];   // [27,1,16]
    const float* W2    = (const float*)d_in[3];   // [27,16,32]
    const float* Wfc   = (const float*)d_in[4];   // [32,10]
    const float* bfc   = (const float*)d_in[5];   // [10]
    float* out = (float*)d_out;                   // [N,10]
    const int N = in_sizes[0];                    // feats element count (Cin=1)

    // workspace layout (halfs)
    _Float16* wsH = (_Float16*)d_ws;
    _Float16* Bw2 = wsH + BW2_OFF;                // 14336 halfs
    _Float16* Bfc = wsH + BFC_OFF;                // 512 halfs
    _Float16* Bw1 = wsH + BW1_OFF;                // 512 halfs
    _Float16* x1  = (_Float16*)((char*)d_ws + 32768);  // (N+1)*16 f16

    const int prepThreads = SENT_OFF + 16;
    prep_kernel<<<(prepThreads + 255) / 256, 256, 0, stream>>>(
        W2, Wfc, W1, Bw2, Bfc, Bw1, x1, N);

    const int tiles16 = (N + 15) / 16;            // l1: one wave per 16 voxels
    l1_kernel<<<(tiles16 + 7) / 8, 256, 0, stream>>>(feats, kmap, Bw1, x1, N);

    const int tiles32 = (N + 31) / 32;            // l2fc: one wave per 32 voxels
    l2fc_kernel<<<(tiles32 + 7) / 8, 256, 0, stream>>>(x1, kmap, Bw2, Bfc, bfc,
                                                       out, N);
}
